// SyntaxMaskGenerator_23390391894230
// MI455X (gfx1250) — matrix operand order repaired and checked by reading, not compiled, not run on hardware
//
#include <hip/hip_runtime.h>

#define B_DIM 4
#define S_DIM 2048
#define H_DIM 4096
#define F_DIM 128
#define ROWS (B_DIM * S_DIM)
#define BSTRIDE 20   // u32 stride per column in LDS B buffer (80B: 16B-aligned, bank-clean)

typedef __attribute__((ext_vector_type(16))) __bf16 v16bf;
typedef __attribute__((ext_vector_type(8)))  float v8f;

union ABf16 { v16bf v; unsigned int u[8]; };

__device__ __forceinline__ unsigned int bf16rne(float x) {
    unsigned int u = __float_as_uint(x);
    unsigned int r = u + 0x7FFFu + ((u >> 16) & 1u);
    return r >> 16;
}
__device__ __forceinline__ unsigned int pack2(float lo, float hi) {
    return bf16rne(lo) | (bf16rne(hi) << 16);
}

// ---------------------------------------------------------------------------
// Kernel A: d[b,s] = relu(X @ Wb1 + bb1) . (Wb2[:,0]-Wb2[:,1]) + (bb2[0]-bb2[1])
// 128 threads (4 waves), 64 rows/block. Wave: 16 rows x 128 cols, K-step 32.
// Wb1 chunks staged to double-buffered LDS as packed bf16 in WMMA B layout;
// B fragments read as 2x ds_read_b128 per 16-col tile.
// ---------------------------------------------------------------------------
__global__ __launch_bounds__(128) void gemm_boundary_kernel(
    const float* __restrict__ X, const float* __restrict__ Wb1,
    const float* __restrict__ bb1, const float* __restrict__ Wb2,
    const float* __restrict__ bb2, float* __restrict__ dout)
{
    __shared__ __align__(16) unsigned int ldsB[2][F_DIM * BSTRIDE];

    const int t    = threadIdx.x;
    const int wave = t >> 5;
    const int lane = t & 31;
    const int mrow = lane & 15;     // A: row-in-tile / B,C: col-in-tile
    const int lh   = lane >> 4;     // lane half
    const int rowBase = blockIdx.x * 64 + wave * 16;
    const size_t xbase = (size_t)(rowBase + mrow) * H_DIM;
    const int kb       = lh ? 8 : 0;   // A K-offset within 32-chunk
    const int pairBase = lh ? 4 : 0;   // B packed-pair offset (K 8..15 / 24..31)

    // staging decomposition: cols 4*cg..4*cg+3, K-pairs pq*4..pq*4+3
    const int cg = t & 31;
    const int pq = t >> 5;

    v8f acc[8] = {};

    for (int kc = 0; kc < H_DIM / 32; kc++) {
        const int k0 = kc * 32;
        unsigned int* buf = ldsB[kc & 1];

        // ---- stage 32x128 chunk of Wb1 -> bf16 packed pairs ----
        #pragma unroll
        for (int u = 0; u < 4; u++) {
            int pp = pq * 4 + u;                      // K-pair 0..15
            const float4 r0 = *((const float4*)(Wb1 + (size_t)(k0 + 2 * pp)     * F_DIM + 4 * cg));
            const float4 r1 = *((const float4*)(Wb1 + (size_t)(k0 + 2 * pp + 1) * F_DIM + 4 * cg));
            buf[(4 * cg + 0) * BSTRIDE + pp] = pack2(r0.x, r1.x);
            buf[(4 * cg + 1) * BSTRIDE + pp] = pack2(r0.y, r1.y);
            buf[(4 * cg + 2) * BSTRIDE + pp] = pack2(r0.z, r1.z);
            buf[(4 * cg + 3) * BSTRIDE + pp] = pack2(r0.w, r1.w);
        }
        __syncthreads();   // single barrier: double-buffering covers WAR on buf

        // ---- A fragment: 16x32 bf16, ISA lane layout ----
        ABf16 a;
        float4 l0 = *((const float4*)(X + xbase + k0 + kb));
        float4 l1 = *((const float4*)(X + xbase + k0 + kb + 4));
        float4 l2 = *((const float4*)(X + xbase + k0 + kb + 16));
        float4 l3 = *((const float4*)(X + xbase + k0 + kb + 20));
        a.u[0] = pack2(l0.x, l0.y); a.u[1] = pack2(l0.z, l0.w);
        a.u[2] = pack2(l1.x, l1.y); a.u[3] = pack2(l1.z, l1.w);
        a.u[4] = pack2(l2.x, l2.y); a.u[5] = pack2(l2.z, l2.w);
        a.u[6] = pack2(l3.x, l3.y); a.u[7] = pack2(l3.z, l3.w);

        // ---- 8 column tiles of 16: 2x ds_read_b128 + wmma each ----
        #pragma unroll
        for (int nt = 0; nt < 8; nt++) {
            const uint4* bp = (const uint4*)&buf[(nt * 16 + mrow) * BSTRIDE + pairBase];
            uint4 x0 = bp[0];
            uint4 x1 = bp[2];
            ABf16 b;
            b.u[0] = x0.x; b.u[1] = x0.y; b.u[2] = x0.z; b.u[3] = x0.w;
            b.u[4] = x1.x; b.u[5] = x1.y; b.u[6] = x1.z; b.u[7] = x1.w;
            acc[nt] = __builtin_amdgcn_wmma_f32_16x16x32_bf16(
                false, a.v, false, b.v, (short)0, acc[nt], false, false);
        }
    }

    // ---- epilogue: bias + relu, dot with (Wb2 col0 - col1), lane reduce ----
    const float bb2d = bb2[0] - bb2[1];
    float p[8];
    #pragma unroll
    for (int r = 0; r < 8; r++) p[r] = 0.0f;
    #pragma unroll
    for (int nt = 0; nt < 8; nt++) {
        int col    = nt * 16 + mrow;
        float bias = bb1[col];
        float w2d  = Wb2[2 * col] - Wb2[2 * col + 1];
        #pragma unroll
        for (int r = 0; r < 8; r++) {
            float h = acc[nt][r] + bias;
            h = h > 0.0f ? h : 0.0f;
            p[r] += h * w2d;
        }
    }
    #pragma unroll
    for (int r = 0; r < 8; r++) {
        #pragma unroll
        for (int m = 1; m < 16; m <<= 1) p[r] += __shfl_xor(p[r], m, 32);
    }
    if (mrow == 0) {                 // lanes 0 (rows 0..7) and 16 (rows 8..15)
        #pragma unroll
        for (int r = 0; r < 8; r++)
            dout[rowBase + lh * 8 + r] = p[r] + bb2d;
    }
}

// ---------------------------------------------------------------------------
// Kernel B: per-batch start/end matching + cummax  ->  f[b][s]
// One 256-thread block per batch, 8 elements/thread, LDS scans.
// ---------------------------------------------------------------------------
__global__ __launch_bounds__(256) void boundary_match_kernel(
    const float* __restrict__ dval, int* __restrict__ fout)
{
    __shared__ int sArr[S_DIM];
    __shared__ int eArr[S_DIM];
    __shared__ int gbuf[S_DIM];
    __shared__ int scanS[256];
    __shared__ int scanE[256];
    __shared__ int scanM[256];

    const int b = blockIdx.x;
    const int t = threadIdx.x;
    const int base = t * 8;
    const float* dv = dval + b * S_DIM;

    int isS[8], isE[8], locS[8], locE[8];
    int cs = 0, ce = 0;
    #pragma unroll
    for (int i = 0; i < 8; i++) {
        float d = dv[base + i];
        isS[i] = d > 0.0f ? 1 : 0;
        isE[i] = d < 0.0f ? 1 : 0;
        cs += isS[i]; locS[i] = cs;
        ce += isE[i]; locE[i] = ce;
        gbuf[base + i] = -1;
    }
    scanS[t] = cs; scanE[t] = ce;
    __syncthreads();
    for (int off = 1; off < 256; off <<= 1) {       // inclusive sum scan
        int vs = scanS[t], ve = scanE[t];
        int as = (t >= off) ? scanS[t - off] : 0;
        int ae = (t >= off) ? scanE[t - off] : 0;
        __syncthreads();
        scanS[t] = vs + as; scanE[t] = ve + ae;
        __syncthreads();
    }
    const int offS = (t > 0) ? scanS[t - 1] : 0;
    const int offE = (t > 0) ? scanE[t - 1] : 0;
    const int totS = scanS[255], totE = scanE[255];
    const int n_pairs = totS < totE ? totS : totE;

    #pragma unroll
    for (int i = 0; i < 8; i++) {                   // scatter ranks -> pos
        if (isS[i]) sArr[offS + locS[i] - 1] = base + i;
        if (isE[i]) eArr[offE + locE[i] - 1] = base + i;
    }
    __syncthreads();
    #pragma unroll
    for (int i = 0; i < 8; i++) {                   // valid pairs -> g[start]=end
        int k = base + i;
        if (k < n_pairs) {
            int s = sArr[k], e = eArr[k];
            if (s < e) gbuf[s] = e;                 // start positions distinct
        }
    }
    __syncthreads();
    int g[8]; int mx = -2147483647;                 // cummax
    #pragma unroll
    for (int i = 0; i < 8; i++) {
        int v = gbuf[base + i];
        mx = v > mx ? v : mx;
        g[i] = mx;
    }
    scanM[t] = mx;
    __syncthreads();
    for (int off = 1; off < 256; off <<= 1) {       // inclusive max scan
        int v = scanM[t];
        int a = (t >= off) ? scanM[t - off] : -2147483647;
        __syncthreads();
        scanM[t] = v > a ? v : a;
        __syncthreads();
    }
    const int pre = (t > 0) ? scanM[t - 1] : -2147483647;
    #pragma unroll
    for (int i = 0; i < 8; i++) {
        int fv = g[i] > pre ? g[i] : pre;
        fout[b * S_DIM + base + i] = fv;
    }
}

// ---------------------------------------------------------------------------
// Kernel C: out[b,i,j] = (max(i,j) < f[b][min(i,j)] ? 2 : 1) * base_mask[b,j]
// One block per (i,b), 512 threads x float4. Pure bandwidth (67 MB store).
// ---------------------------------------------------------------------------
__global__ __launch_bounds__(512) void mask_write_kernel(
    const int* __restrict__ f, const float* __restrict__ base_mask,
    float* __restrict__ out)
{
    const int b = blockIdx.y;
    const int i = blockIdx.x;
    const int j0 = threadIdx.x * 4;
    const int fi = f[b * S_DIM + i];
    int4   fj = *((const int4*)  (f         + b * S_DIM + j0));
    float4 bm = *((const float4*)(base_mask + b * S_DIM + j0));
    float4 r;
    int j;
    j = j0 + 0; r.x = (((j < i) ? (i < fj.x) : (j < fi)) ? 2.0f : 1.0f) * bm.x;
    j = j0 + 1; r.y = (((j < i) ? (i < fj.y) : (j < fi)) ? 2.0f : 1.0f) * bm.y;
    j = j0 + 2; r.z = (((j < i) ? (i < fj.z) : (j < fi)) ? 2.0f : 1.0f) * bm.z;
    j = j0 + 3; r.w = (((j < i) ? (i < fj.w) : (j < fi)) ? 2.0f : 1.0f) * bm.w;
    size_t o = ((size_t)b * S_DIM + i) * S_DIM + j0;
    *((float4*)(out + o)) = r;
}

extern "C" void kernel_launch(void* const* d_in, const int* in_sizes, int n_in,
                              void* d_out, int out_size, void* d_ws, size_t ws_size,
                              hipStream_t stream) {
    // Input order: hidden_states, base_mask, W1, b1, W2, b2, Wb1, bb1, Wb2, bb2
    // (W1/b1/W2/b2 path is dead code in the reference -> skipped entirely.)
    const float* X         = (const float*)d_in[0];
    const float* base_mask = (const float*)d_in[1];
    const float* Wb1       = (const float*)d_in[6];
    const float* bb1       = (const float*)d_in[7];
    const float* Wb2       = (const float*)d_in[8];
    const float* bb2       = (const float*)d_in[9];
    float* out = (float*)d_out;

    float* dscr = (float*)d_ws;                                   // 8192 floats
    int*   fscr = (int*)((char*)d_ws + ROWS * sizeof(float));     // 8192 ints

    gemm_boundary_kernel<<<ROWS / 64, 128, 0, stream>>>(X, Wb1, bb1, Wb2, bb2, dscr);
    boundary_match_kernel<<<B_DIM, 256, 0, stream>>>(dscr, fscr);
    mask_write_kernel<<<dim3(S_DIM, B_DIM), 512, 0, stream>>>(fscr, base_mask, out);
}
